// Transformer_32049045963356
// MI455X (gfx1250) — matrix operand order repaired and checked by reading, not compiled, not run on hardware
//
#include <hip/hip_runtime.h>
#include <hip/hip_bf16.h>

// ---------------- model constants ----------------
#define DIM 2048
#define N_LAYERS 4
#define N_HEADS 16
#define N_KV_HEADS 8
#define HEAD_DIM 128
#define HIDDEN 5632
#define VOCAB 32000
#define SEQ 2048
#define EPS 1e-5f
#define QD (N_HEADS * HEAD_DIM)     // 2048
#define KVD (N_KV_HEADS * HEAD_DIM) // 1024

// ---------------- WMMA types ----------------
typedef __attribute__((ext_vector_type(16))) __bf16 v16bf;
typedef __attribute__((ext_vector_type(8)))  float  v8f;

union Frag16 { v16bf v; uint4 q[2]; };

// ---------------- helpers ----------------
__device__ __forceinline__ unsigned short f32_to_bf16(float f) {
    unsigned int u = __float_as_uint(f);
    if ((u & 0x7fffffffu) > 0x7f800000u) return (unsigned short)((u >> 16) | 0x40u); // quiet NaN
    unsigned int r = u + 0x7fffu + ((u >> 16) & 1u); // round-to-nearest-even
    return (unsigned short)(r >> 16);
}

// ---------------- fp32 -> bf16 conversion, vectorized (grid-stride) ----------------
__global__ void cvt_bf16_kernel(const float4* __restrict__ in, ushort4* __restrict__ out, long n4) {
    long i = (long)blockIdx.x * blockDim.x + threadIdx.x;
    long stride = (long)gridDim.x * blockDim.x;
    for (; i < n4; i += stride) {
        float4 f = in[i];
        ushort4 o;
        o.x = f32_to_bf16(f.x); o.y = f32_to_bf16(f.y);
        o.z = f32_to_bf16(f.z); o.w = f32_to_bf16(f.w);
        out[i] = o;
    }
}

// ---------------- embedding gather ----------------
__global__ void embed_kernel(const int* __restrict__ ids, const float* __restrict__ emb,
                             float* __restrict__ h) {
    int s = blockIdx.x;
    long row = (long)ids[s] * DIM;
    for (int i = threadIdx.x; i < DIM; i += blockDim.x)
        h[(long)s * DIM + i] = emb[row + i];
}

// ---------------- RMSNorm (f32 in, bf16 out) ----------------
__global__ __launch_bounds__(256) void rmsnorm_kernel(const float* __restrict__ h,
                                                      const float* __restrict__ w,
                                                      unsigned short* __restrict__ xb) {
    __shared__ float red[256];
    int s = blockIdx.x, tid = threadIdx.x;
    const float* row = h + (long)s * DIM;
    float ss = 0.f;
    for (int i = tid; i < DIM; i += 256) { float v = row[i]; ss += v * v; }
    red[tid] = ss; __syncthreads();
    for (int off = 128; off > 0; off >>= 1) {
        if (tid < off) red[tid] += red[tid + off];
        __syncthreads();
    }
    float inv = rsqrtf(red[0] / (float)DIM + EPS);
    for (int i = tid; i < DIM; i += 256)
        xb[(long)s * DIM + i] = f32_to_bf16(row[i] * inv * w[i]);
}

// ---------------- RoPE on q (16 heads) and k (8 heads), in-place f32 ----------------
__global__ void rope_kernel(float* __restrict__ q, float* __restrict__ k) {
    int s = blockIdx.x, hh = blockIdx.y, i = threadIdx.x; // i: pair index 0..63
    float inv = __powf(10000.0f, -(float)(2 * i) / (float)HEAD_DIM);
    float ang = (float)s * inv;
    float c = cosf(ang), sn = sinf(ang);
    float* base = (hh < N_HEADS) ? (q + (long)s * QD + hh * HEAD_DIM)
                                 : (k + (long)s * KVD + (hh - N_HEADS) * HEAD_DIM);
    float xr = base[2 * i], xi = base[2 * i + 1];
    base[2 * i]     = xr * c - xi * sn;
    base[2 * i + 1] = xr * sn + xi * c;
}

// ---------------- WMMA flash attention ----------------
// Grid (SEQ/128, N_HEADS), 256 threads = 8 waves. Wave w owns query rows
// [B0 + 16w, B0 + 16w + 15]; workgroup shares K/V key blocks of 32 in LDS.
// S = Q.K^T : 8 WMMAs/keyblock (4 d-chunks x 2 key tiles), online softmax in
// C-register layout (row = 16 lanes of a lane-half -> 4x shfl_xor reductions),
// P routed through LDS to A-fragment layout, O += P.V : 4 WMMAs/keyblock.
__global__ __launch_bounds__(256) void fattn_kernel(const unsigned short* __restrict__ qb,
                                                    const unsigned short* __restrict__ kb,
                                                    const unsigned short* __restrict__ vb,
                                                    unsigned short* __restrict__ yb) {
    __shared__ unsigned short Ks[32 * 128];   // [key][d]
    __shared__ unsigned short VsT[128 * 32];  // [d][key]
    __shared__ unsigned short Ps[8][16 * 32]; // per-wave P tile [q][key]

    int tid = threadIdx.x, wid = tid >> 5, lane = tid & 31;
    int half = lane >> 4, l16 = lane & 15;
    int h = blockIdx.y, kvh = h >> 1;
    int B0 = blockIdx.x * 128;
    int qBase = B0 + wid * 16;
    int qmax = qBase + 15;
    const float scale = 0.08838834764831845f; // 1/sqrt(128)

    // persistent Q A-fragments (4 chunks of d=32)
    Frag16 aq[4];
    {
        const unsigned short* qp = qb + (unsigned)(qBase + l16) * QD + h * HEAD_DIM;
        #pragma unroll
        for (int dc = 0; dc < 4; ++dc) {
            aq[dc].q[0] = *(const uint4*)(qp + dc * 32 + half * 16);
            aq[dc].q[1] = *(const uint4*)(qp + dc * 32 + half * 16 + 8);
        }
    }

    v8f O[4] = {};
    float m[8], l[8];
    #pragma unroll
    for (int r = 0; r < 8; ++r) { m[r] = -3.0e38f; l[r] = 0.f; }

    // cooperative K/V load assignments
    int krow = tid >> 3;          // 0..31
    int kcol = (tid & 7) * 16;    // 0..112
    int vk   = (tid & 7) * 4;     // key group of 4 (tid<128)
    int vd   = (tid >> 3) * 8;    // d chunk of 8  (tid<128 -> 0..120)

    int nkb = B0 / 32 + 4;
    for (int t = 0; t < nkb; ++t) {
        int k0 = t * 32;
        __syncthreads();
        { // K: straight copy [key][d]
            const unsigned short* kp = kb + (unsigned)(k0 + krow) * KVD + kvh * HEAD_DIM + kcol;
            *(uint4*)(&Ks[krow * 128 + kcol])     = *(const uint4*)(kp);
            *(uint4*)(&Ks[krow * 128 + kcol + 8]) = *(const uint4*)(kp + 8);
        }
        if (tid < 128) { // V: transposed [d][key], ds_store_b64 packing
            const unsigned short* vp = vb + (unsigned)(k0 + vk) * KVD + kvh * HEAD_DIM + vd;
            uint4 r0 = *(const uint4*)(vp);
            uint4 r1 = *(const uint4*)(vp + KVD);
            uint4 r2 = *(const uint4*)(vp + 2 * KVD);
            uint4 r3 = *(const uint4*)(vp + 3 * KVD);
            const unsigned short* p0 = (const unsigned short*)&r0;
            const unsigned short* p1 = (const unsigned short*)&r1;
            const unsigned short* p2 = (const unsigned short*)&r2;
            const unsigned short* p3 = (const unsigned short*)&r3;
            #pragma unroll
            for (int i = 0; i < 8; ++i) {
                ushort4 w;
                w.x = p0[i]; w.y = p1[i]; w.z = p2[i]; w.w = p3[i];
                *(ushort4*)(&VsT[(vd + i) * 32 + vk]) = w;
            }
        }
        __syncthreads();
        if (k0 > qmax) continue; // wave-uniform causal skip (EXEC stays all-ones)

        // S = Q . K^T over two 16-key tiles
        v8f s0 = {}, s1 = {};
        #pragma unroll
        for (int dc = 0; dc < 4; ++dc) {
            Frag16 b0, b1;
            b0.q[0] = *(const uint4*)(&Ks[l16 * 128 + dc * 32 + half * 16]);
            b0.q[1] = *(const uint4*)(&Ks[l16 * 128 + dc * 32 + half * 16 + 8]);
            b1.q[0] = *(const uint4*)(&Ks[(16 + l16) * 128 + dc * 32 + half * 16]);
            b1.q[1] = *(const uint4*)(&Ks[(16 + l16) * 128 + dc * 32 + half * 16 + 8]);
            s0 = __builtin_amdgcn_wmma_f32_16x16x32_bf16(false, aq[dc].v, false, b0.v, (short)0, s0, false, false);
            s1 = __builtin_amdgcn_wmma_f32_16x16x32_bf16(false, aq[dc].v, false, b1.v, (short)0, s1, false, false);
        }

        // online softmax; C-layout row = qBase + half*8 + r across 16 lanes
        #pragma unroll
        for (int r = 0; r < 8; ++r) {
            int qrow = qBase + half * 8 + r;
            float v0 = (k0 + l16      <= qrow) ? s0[r] * scale : -3.0e38f;
            float v1 = (k0 + 16 + l16 <= qrow) ? s1[r] * scale : -3.0e38f;
            float mr = fmaxf(v0, v1);
            mr = fmaxf(mr, __shfl_xor(mr, 1, 32));
            mr = fmaxf(mr, __shfl_xor(mr, 2, 32));
            mr = fmaxf(mr, __shfl_xor(mr, 4, 32));
            mr = fmaxf(mr, __shfl_xor(mr, 8, 32));
            float mnew = fmaxf(m[r], mr);
            float es = __expf(m[r] - mnew);
            m[r] = mnew;
            float p0 = __expf(v0 - mnew);
            float p1 = __expf(v1 - mnew);
            float rs = p0 + p1;
            rs += __shfl_xor(rs, 1, 32);
            rs += __shfl_xor(rs, 2, 32);
            rs += __shfl_xor(rs, 4, 32);
            rs += __shfl_xor(rs, 8, 32);
            l[r] = l[r] * es + rs;
            #pragma unroll
            for (int dc = 0; dc < 4; ++dc) O[dc][r] *= es;
            int prow = half * 8 + r;
            Ps[wid][prow * 32 + l16]      = f32_to_bf16(p0);
            Ps[wid][prow * 32 + 16 + l16] = f32_to_bf16(p1);
        }
        asm volatile("s_wait_dscnt 0x0" ::: "memory"); // intra-wave DS RAW: P stores -> A-frag loads

        // O += P . V
        Frag16 pa;
        pa.q[0] = *(const uint4*)(&Ps[wid][l16 * 32 + half * 16]);
        pa.q[1] = *(const uint4*)(&Ps[wid][l16 * 32 + half * 16 + 8]);
        #pragma unroll
        for (int dc = 0; dc < 4; ++dc) {
            Frag16 bv;
            bv.q[0] = *(const uint4*)(&VsT[(dc * 16 + l16) * 32 + half * 16]);
            bv.q[1] = *(const uint4*)(&VsT[(dc * 16 + l16) * 32 + half * 16 + 8]);
            O[dc] = __builtin_amdgcn_wmma_f32_16x16x32_bf16(false, pa.v, false, bv.v, (short)0, O[dc], false, false);
        }
    }

    // normalize and store (bf16 for the wo GEMM)
    #pragma unroll
    for (int r = 0; r < 8; ++r) {
        int qrow = qBase + half * 8 + r;
        float invl = 1.0f / l[r];
        #pragma unroll
        for (int dc = 0; dc < 4; ++dc)
            yb[(unsigned)qrow * QD + h * HEAD_DIM + dc * 16 + l16] = f32_to_bf16(O[dc][r] * invl);
    }
}

// ---------------- SiLU(a1) * a3 -> bf16 ----------------
__global__ void silu_mul_kernel(const float* __restrict__ a1, const float* __restrict__ a3,
                                unsigned short* __restrict__ g, long n) {
    long i = (long)blockIdx.x * blockDim.x + threadIdx.x;
    long stride = (long)gridDim.x * blockDim.x;
    for (; i < n; i += stride) {
        float x = a1[i];
        float sv = x / (1.f + __expf(-x));
        g[i] = f32_to_bf16(sv * a3[i]);
    }
}

// ---------------- bf16 WMMA GEMM: C[M,N] (f32) = A[M,K](bf16) @ B[K,N](bf16) (+ res) ----
#define BM 64
#define BN 256
#define BK 32

template <bool RES>
__global__ __launch_bounds__(256) void gemm_bf16_kernel(const unsigned short* __restrict__ A,
                                                        const unsigned short* __restrict__ B,
                                                        const float* Rp, float* C,
                                                        int M, int N, int K) {
    __shared__ unsigned short As[2][BM * BK];   // [m][k]
    __shared__ unsigned short BsT[2][BN * BK];  // [n][k]

    int tid  = threadIdx.x;
    int wid  = tid >> 5;
    int lane = tid & 31;
    int half = lane >> 4;
    int l16  = lane & 15;

    unsigned mTile = blockIdx.y * BM;
    unsigned nTile = blockIdx.x * BN;
    int waveM = (wid & 1) * 32;
    int waveN = (wid >> 1) * 64;

    v8f acc[2][4] = {};

    int ar  = tid >> 2;
    int ak  = (tid & 3) * 8;
    int bk0 = (tid >> 5) * 4;
    int bn0 = (tid & 31) * 8;

    const unsigned short* Aptr  = A + (unsigned)(mTile + ar) * (unsigned)K + (unsigned)ak;
    const unsigned short* Bbase = B + nTile + (unsigned)bn0;

    uint4 av  = *(const uint4*)(Aptr);
    uint4 bv0 = *(const uint4*)(Bbase + (unsigned)(bk0 + 0) * (unsigned)N);
    uint4 bv1 = *(const uint4*)(Bbase + (unsigned)(bk0 + 1) * (unsigned)N);
    uint4 bv2 = *(const uint4*)(Bbase + (unsigned)(bk0 + 2) * (unsigned)N);
    uint4 bv3 = *(const uint4*)(Bbase + (unsigned)(bk0 + 3) * (unsigned)N);

    int nk = K / BK;
    int buf = 0;
    for (int t = 0; t < nk; ++t) {
        *(uint4*)(&As[buf][ar * BK + ak]) = av;
        {
            const unsigned short* p0 = (const unsigned short*)&bv0;
            const unsigned short* p1 = (const unsigned short*)&bv1;
            const unsigned short* p2 = (const unsigned short*)&bv2;
            const unsigned short* p3 = (const unsigned short*)&bv3;
            #pragma unroll
            for (int i = 0; i < 8; ++i) {
                ushort4 w;
                w.x = p0[i]; w.y = p1[i]; w.z = p2[i]; w.w = p3[i];
                *(ushort4*)(&BsT[buf][(bn0 + i) * BK + bk0]) = w;
            }
        }
        __syncthreads();

        if (t + 1 < nk) {
            unsigned koff = (unsigned)(t + 1) * BK;
            av  = *(const uint4*)(Aptr + koff);
            bv0 = *(const uint4*)(Bbase + (koff + bk0 + 0) * (unsigned)N);
            bv1 = *(const uint4*)(Bbase + (koff + bk0 + 1) * (unsigned)N);
            bv2 = *(const uint4*)(Bbase + (koff + bk0 + 2) * (unsigned)N);
            bv3 = *(const uint4*)(Bbase + (koff + bk0 + 3) * (unsigned)N);
        }

        Frag16 a[2], b[4];
        #pragma unroll
        for (int mi = 0; mi < 2; ++mi) {
            int row = waveM + mi * 16 + l16;
            a[mi].q[0] = *(const uint4*)(&As[buf][row * BK + half * 16]);
            a[mi].q[1] = *(const uint4*)(&As[buf][row * BK + half * 16 + 8]);
        }
        #pragma unroll
        for (int ni = 0; ni < 4; ++ni) {
            int col = waveN + ni * 16 + l16;
            b[ni].q[0] = *(const uint4*)(&BsT[buf][col * BK + half * 16]);
            b[ni].q[1] = *(const uint4*)(&BsT[buf][col * BK + half * 16 + 8]);
        }
        #pragma unroll
        for (int mi = 0; mi < 2; ++mi)
            #pragma unroll
            for (int ni = 0; ni < 4; ++ni)
                acc[mi][ni] = __builtin_amdgcn_wmma_f32_16x16x32_bf16(
                    false, a[mi].v, false, b[ni].v, (short)0, acc[mi][ni], false, false);

        buf ^= 1;
    }

    unsigned base = (mTile + (unsigned)(waveM + half * 8)) * (unsigned)N
                  + nTile + (unsigned)(waveN + l16);
    #pragma unroll
    for (int mi = 0; mi < 2; ++mi) {
        #pragma unroll
        for (int ni = 0; ni < 4; ++ni) {
            #pragma unroll
            for (int r = 0; r < 8; ++r) {
                unsigned idx = base + (unsigned)(mi * 16 + r) * (unsigned)N + (unsigned)(ni * 16);
                float val = acc[mi][ni][r];
                if (RES) val += Rp[idx];
                C[idx] = val;
            }
        }
    }
}

// ---------------- launch ----------------
static inline void* bump(char*& p, size_t bytes) {
    void* r = (void*)p;
    p += (bytes + 255) & ~(size_t)255;
    return r;
}

extern "C" void kernel_launch(void* const* d_in, const int* in_sizes, int n_in,
                              void* d_out, int out_size, void* d_ws, size_t ws_size,
                              hipStream_t stream) {
    (void)in_sizes; (void)n_in; (void)out_size; (void)ws_size;

    const int*   ids  = (const int*)d_in[0];
    const float* emb  = (const float*)d_in[1];
    const float* wq   = (const float*)d_in[2];
    const float* wk   = (const float*)d_in[3];
    const float* wv   = (const float*)d_in[4];
    const float* wo   = (const float*)d_in[5];
    const float* w1   = (const float*)d_in[6];
    const float* w2   = (const float*)d_in[7];
    const float* w3   = (const float*)d_in[8];
    const float* anw  = (const float*)d_in[9];
    const float* fnw  = (const float*)d_in[10];
    const float* flnw = (const float*)d_in[11];
    const float* ow   = (const float*)d_in[12];
    float* out = (float*)d_out;

    char* p = (char*)d_ws;
    const long nWq = (long)N_LAYERS * DIM * QD;
    const long nWk = (long)N_LAYERS * DIM * KVD;
    const long nWv = nWk;
    const long nWo = (long)N_LAYERS * QD * DIM;
    const long nW1 = (long)N_LAYERS * DIM * HIDDEN;
    const long nW2 = (long)N_LAYERS * HIDDEN * DIM;
    const long nW3 = nW1;
    const long nOw = (long)DIM * VOCAB;

    unsigned short* wq_b = (unsigned short*)bump(p, nWq * 2);
    unsigned short* wk_b = (unsigned short*)bump(p, nWk * 2);
    unsigned short* wv_b = (unsigned short*)bump(p, nWv * 2);
    unsigned short* wo_b = (unsigned short*)bump(p, nWo * 2);
    unsigned short* w1_b = (unsigned short*)bump(p, nW1 * 2);
    unsigned short* w2_b = (unsigned short*)bump(p, nW2 * 2);
    unsigned short* w3_b = (unsigned short*)bump(p, nW3 * 2);
    unsigned short* ow_b = (unsigned short*)bump(p, nOw * 2);

    float*          h   = (float*)bump(p, (size_t)SEQ * DIM * 4);
    unsigned short* xb  = (unsigned short*)bump(p, (size_t)SEQ * DIM * 2);
    float*          qf  = (float*)bump(p, (size_t)SEQ * QD * 4);
    float*          kf  = (float*)bump(p, (size_t)SEQ * KVD * 4);
    float*          vf  = (float*)bump(p, (size_t)SEQ * KVD * 4);
    unsigned short* qbb = (unsigned short*)bump(p, (size_t)SEQ * QD * 2);
    unsigned short* kbb = (unsigned short*)bump(p, (size_t)SEQ * KVD * 2);
    unsigned short* vbb = (unsigned short*)bump(p, (size_t)SEQ * KVD * 2);
    unsigned short* yb  = (unsigned short*)bump(p, (size_t)SEQ * QD * 2);
    float*          a1  = (float*)bump(p, (size_t)SEQ * HIDDEN * 4);
    float*          a3  = (float*)bump(p, (size_t)SEQ * HIDDEN * 4);
    unsigned short* gb  = (unsigned short*)bump(p, (size_t)SEQ * HIDDEN * 2);

    // weight conversion (fp32 -> bf16): halves the dominant HBM stream
    cvt_bf16_kernel<<<4096, 256, 0, stream>>>((const float4*)wq, (ushort4*)wq_b, nWq / 4);
    cvt_bf16_kernel<<<4096, 256, 0, stream>>>((const float4*)wk, (ushort4*)wk_b, nWk / 4);
    cvt_bf16_kernel<<<4096, 256, 0, stream>>>((const float4*)wv, (ushort4*)wv_b, nWv / 4);
    cvt_bf16_kernel<<<4096, 256, 0, stream>>>((const float4*)wo, (ushort4*)wo_b, nWo / 4);
    cvt_bf16_kernel<<<4096, 256, 0, stream>>>((const float4*)w1, (ushort4*)w1_b, nW1 / 4);
    cvt_bf16_kernel<<<4096, 256, 0, stream>>>((const float4*)w2, (ushort4*)w2_b, nW2 / 4);
    cvt_bf16_kernel<<<4096, 256, 0, stream>>>((const float4*)w3, (ushort4*)w3_b, nW3 / 4);
    cvt_bf16_kernel<<<4096, 256, 0, stream>>>((const float4*)ow, (ushort4*)ow_b, nOw / 4);

    embed_kernel<<<SEQ, 256, 0, stream>>>(ids, emb, h);

    for (int l = 0; l < N_LAYERS; ++l) {
        rmsnorm_kernel<<<SEQ, 256, 0, stream>>>(h, anw + (long)l * DIM, xb);

        gemm_bf16_kernel<false><<<dim3(QD / BN, SEQ / BM), 256, 0, stream>>>(
            xb, wq_b + (long)l * DIM * QD, nullptr, qf, SEQ, QD, DIM);
        gemm_bf16_kernel<false><<<dim3(KVD / BN, SEQ / BM), 256, 0, stream>>>(
            xb, wk_b + (long)l * DIM * KVD, nullptr, kf, SEQ, KVD, DIM);
        gemm_bf16_kernel<false><<<dim3(KVD / BN, SEQ / BM), 256, 0, stream>>>(
            xb, wv_b + (long)l * DIM * KVD, nullptr, vf, SEQ, KVD, DIM);

        rope_kernel<<<dim3(SEQ, N_HEADS + N_KV_HEADS), 64, 0, stream>>>(qf, kf);

        cvt_bf16_kernel<<<2048, 256, 0, stream>>>((const float4*)qf, (ushort4*)qbb, (long)SEQ * QD / 4);
        cvt_bf16_kernel<<<2048, 256, 0, stream>>>((const float4*)kf, (ushort4*)kbb, (long)SEQ * KVD / 4);
        cvt_bf16_kernel<<<2048, 256, 0, stream>>>((const float4*)vf, (ushort4*)vbb, (long)SEQ * KVD / 4);

        fattn_kernel<<<dim3(SEQ / 128, N_HEADS), 256, 0, stream>>>(qbb, kbb, vbb, yb);

        gemm_bf16_kernel<true><<<dim3(DIM / BN, SEQ / BM), 256, 0, stream>>>(
            yb, wo_b + (long)l * QD * DIM, h, h, SEQ, DIM, QD);

        rmsnorm_kernel<<<SEQ, 256, 0, stream>>>(h, fnw + (long)l * DIM, xb);

        gemm_bf16_kernel<false><<<dim3(HIDDEN / BN, SEQ / BM), 256, 0, stream>>>(
            xb, w1_b + (long)l * DIM * HIDDEN, nullptr, a1, SEQ, HIDDEN, DIM);
        gemm_bf16_kernel<false><<<dim3(HIDDEN / BN, SEQ / BM), 256, 0, stream>>>(
            xb, w3_b + (long)l * DIM * HIDDEN, nullptr, a3, SEQ, HIDDEN, DIM);

        silu_mul_kernel<<<4096, 256, 0, stream>>>(a1, a3, gb, (long)SEQ * HIDDEN);

        gemm_bf16_kernel<true><<<dim3(DIM / BN, SEQ / BM), 256, 0, stream>>>(
            gb, w2_b + (long)l * HIDDEN * DIM, h, h, SEQ, DIM, HIDDEN);
    }

    rmsnorm_kernel<<<SEQ, 256, 0, stream>>>(h, flnw, xb);
    gemm_bf16_kernel<false><<<dim3(VOCAB / BN, SEQ / BM), 256, 0, stream>>>(
        xb, ow_b, nullptr, out, SEQ, VOCAB, DIM);
}
